// CaptionModel_16475494547468
// MI455X (gfx1250) — compile-verified, matrix operand order checked by reading, not run on hardware
//
#include <hip/hip_runtime.h>
#include <math.h>

// ---------------------------------------------------------------------------
// CaptionModel (CNN -> LSTM -> log_softmax) for MI455X / gfx1250, wave32.
// All GEMMs through v_wmma_f32_16x16x32_bf16 (f32 accumulate).
// ---------------------------------------------------------------------------

#define T_LEN 64
#define BATCH 1024
#define VOCAB 128
#define HID   512
#define G4    (4 * HID)
#define FEAT  3600
#define FEATP 3616   // FEAT padded to a multiple of 32 for the K loop

typedef __attribute__((ext_vector_type(16))) __bf16 v16bf;
typedef __attribute__((ext_vector_type(8)))  float  v8f;
typedef unsigned short u16;

union BFVec { uint4 q[2]; v16bf v; };

__device__ __forceinline__ u16 f2bf(float x) {
  union { float f; unsigned u; } v; v.f = x;
  unsigned u = v.u;
  unsigned r = u + 0x7FFFu + ((u >> 16) & 1u);   // round-to-nearest-even
  return (u16)(r >> 16);
}

__device__ __forceinline__ v8f vzero8() {
  v8f z = {0.f,0.f,0.f,0.f,0.f,0.f,0.f,0.f};
  return z;
}

// Per-lane fragment of a 16x32 bf16 tile (rows x K), CDNA5 wave32 WMMA layout:
//   lanes 0-15 : halves 0-7 = K 0-7,  halves 8-15 = K 16-23
//   lanes 16-31: halves 0-7 = K 8-15, halves 8-15 = K 24-31
// For A: lane&15 = matrix row (M). For B with weights pre-transposed to
// [N][K] row-major: lane&15 = matrix column (N). Two 128-bit loads per lane.
__device__ __forceinline__ v16bf load_frag(const u16* __restrict__ base, int ld, int lane) {
  const u16* p = base + (lane & 15) * ld + ((lane >> 4) << 3);
  BFVec u;
  u.q[0] = *(const uint4*)(p);
  u.q[1] = *(const uint4*)(p + 16);
  return u.v;
}

__device__ __forceinline__ v8f wmma_bf16(v16bf a, v16bf b, v8f c) {
  return __builtin_amdgcn_wmma_f32_16x16x32_bf16(false, a, false, b, (short)0, c, false, false);
}

__device__ __forceinline__ float sigmoidf(float x) { return 1.0f / (1.0f + expf(-x)); }

// ---------------------------------------------------------------------------
// Prep kernels: fp32 -> bf16 conversion / transpose+pad
// ---------------------------------------------------------------------------

// Row-major copy-convert: src f32 [R][K] -> dst bf16 [R][K] (K already mult of 32)
__global__ void k_convert_rows(const float* __restrict__ src, u16* __restrict__ dst, int n) {
  int i = blockIdx.x * blockDim.x + threadIdx.x;
  if (i < n) dst[i] = f2bf(src[i]);
}

// Transpose-convert: src f32 [K][N] -> dst bf16 [N][Kpad], zero-padded K tail
__global__ void k_transpose_convert(const float* __restrict__ src, u16* __restrict__ dst,
                                    int K, int N, int Kpad) {
  int i = blockIdx.x * blockDim.x + threadIdx.x;
  if (i >= N * Kpad) return;
  int kp = i % Kpad, n = i / Kpad;
  dst[(size_t)n * Kpad + kp] = (kp < K) ? f2bf(src[(size_t)kp * N + n]) : (u16)0;
}

// ---------------------------------------------------------------------------
// CNN front-end (small fraction of total FLOPs -> plain VALU)
// ---------------------------------------------------------------------------

// conv 3x3 pad1 on [B,1,64,64] -> relu -> maxpool2 -> pool1 [B,8,32,32] f32
__global__ void k_conv1_pool(const float* __restrict__ img, const float* __restrict__ w,
                             const float* __restrict__ b, float* __restrict__ out) {
  int idx = blockIdx.x * blockDim.x + threadIdx.x;
  if (idx >= BATCH * 8 * 32 * 32) return;
  int x = idx & 31, y = (idx >> 5) & 31, c = (idx >> 10) & 7, bb = idx >> 13;
  const float* im = img + (size_t)bb * 64 * 64;
  const float* wc = w + c * 9;
  float bias = b[c];
  float m = -3.4e38f;
  #pragma unroll
  for (int py = 0; py < 2; ++py)
  #pragma unroll
  for (int px = 0; px < 2; ++px) {
    int oy = 2 * y + py, ox = 2 * x + px;
    float s = bias;
    #pragma unroll
    for (int ky = 0; ky < 3; ++ky) {
      int iy = oy - 1 + ky;
      if ((unsigned)iy >= 64u) continue;
      #pragma unroll
      for (int kx = 0; kx < 3; ++kx) {
        int ix = ox - 1 + kx;
        if ((unsigned)ix >= 64u) continue;
        s = fmaf(wc[ky * 3 + kx], im[iy * 64 + ix], s);
      }
    }
    m = fmaxf(m, s);
  }
  out[idx] = fmaxf(m, 0.0f);   // pool(relu(x)) == relu(pool(x))
}

// conv 5x5 pad1 on pool1 [B,8,32,32] -> relu -> maxpool2 -> flatten bf16 [B][FEATP]
__global__ void k_conv2_pool_flatten(const float* __restrict__ in, const float* __restrict__ w,
                                     const float* __restrict__ b, u16* __restrict__ feat) {
  int idx = blockIdx.x * blockDim.x + threadIdx.x;
  if (idx >= BATCH * 16 * 15 * 15) return;
  int x = idx % 15, y = (idx / 15) % 15, c = (idx / 225) % 16, bb = idx / 3600;
  const float* base = in + (size_t)bb * 8 * 32 * 32;
  float m = -3.4e38f;
  for (int py = 0; py < 2; ++py)
  for (int px = 0; px < 2; ++px) {
    int oy = 2 * y + py, ox = 2 * x + px;     // conv output coord in [0,30)
    float s = b[c];
    for (int ci = 0; ci < 8; ++ci) {
      const float* ip = base + (size_t)ci * 32 * 32;
      const float* wp = w + ((c * 8 + ci) * 25);
      #pragma unroll
      for (int ky = 0; ky < 5; ++ky) {
        int iy = oy - 1 + ky;
        if ((unsigned)iy >= 32u) continue;
        #pragma unroll
        for (int kx = 0; kx < 5; ++kx) {
          int ix = ox - 1 + kx;
          if ((unsigned)ix >= 32u) continue;
          s = fmaf(wp[ky * 5 + kx], ip[iy * 32 + ix], s);
        }
      }
    }
    m = fmaxf(m, s);
  }
  feat[(size_t)bb * FEATP + c * 225 + y * 15 + x] = f2bf(fmaxf(m, 0.0f));
}

// zero the K padding of feat [B][FEAT..FEATP)
__global__ void k_zero_featpad(u16* __restrict__ feat) {
  int i = blockIdx.x * blockDim.x + threadIdx.x;
  if (i >= BATCH * (FEATP - FEAT)) return;
  int bb = i / (FEATP - FEAT), j = i % (FEATP - FEAT);
  feat[(size_t)bb * FEATP + FEAT + j] = 0;
}

// ---------------------------------------------------------------------------
// imgfc GEMM: e = relu(feat[B][FEATP] @ Wt[HID][FEATP]^T + bias) -> f32 [B][HID]
// one wave per 16x16 output tile
// ---------------------------------------------------------------------------
__global__ void __launch_bounds__(32)
k_imgfc_gemm(const u16* __restrict__ A, const u16* __restrict__ Wt,
             const float* __restrict__ bias, float* __restrict__ out) {
  const int lane = threadIdx.x;
  const int n0 = blockIdx.x * 16;         // hidden col
  const int m0 = blockIdx.y * 16;         // batch row
  const u16* Arow = A + (size_t)m0 * FEATP;
  const u16* Brow = Wt + (size_t)n0 * FEATP;
  v8f acc = vzero8();
  for (int k = 0; k < FEATP; k += 32) {
    v16bf a = load_frag(Arow + k, FEATP, lane);
    v16bf b = load_frag(Brow + k, FEATP, lane);
    acc = wmma_bf16(a, b, acc);
  }
  const int col = lane & 15, hi = lane >> 4;
  const float bv = bias[n0 + col];
  #pragma unroll
  for (int r = 0; r < 8; ++r)
    out[(size_t)(m0 + 8 * hi + r) * HID + n0 + col] = fmaxf(acc[r] + bv, 0.0f);
}

// ---------------------------------------------------------------------------
// Fused LSTM step: gates = x_t@xh_w + h_{t-1}@hh_w + xh_b + hh_b (+img4 at t=0)
// then cell update. One wave computes the 16x16 tile for ALL FOUR gates so the
// elementwise update is wave-local. hprev==null => t==0 (h0=c0=0, add img emb).
// ---------------------------------------------------------------------------
__global__ void __launch_bounds__(32)
k_lstm_step(const u16* __restrict__ xt,      // [B][V] bf16
            const u16* __restrict__ xh_wT,   // [4H][V] bf16
            const float* __restrict__ xh_b,  // [4H]
            const u16* __restrict__ hprev,   // [B][H] bf16 or null
            const u16* __restrict__ hh_wT,   // [4H][H] bf16
            const float* __restrict__ hh_b,  // [4H]
            const float* __restrict__ eimg,  // [B][H] f32 or null
            float* __restrict__ cstate,      // [B][H] f32 (read t>0, written)
            u16* __restrict__ hout)          // [B][H] bf16 (slice of hs)
{
  const int lane = threadIdx.x;
  const int n0 = blockIdx.x * 16;     // hidden col
  const int m0 = blockIdx.y * 16;     // batch row
  v8f acc[4];
  #pragma unroll
  for (int g = 0; g < 4; ++g) acc[g] = vzero8();

  // x_t contribution, K = VOCAB = 128
  {
    const u16* Arow = xt + (size_t)m0 * VOCAB;
    for (int k = 0; k < VOCAB; k += 32) {
      v16bf a = load_frag(Arow + k, VOCAB, lane);
      #pragma unroll
      for (int g = 0; g < 4; ++g) {
        v16bf b = load_frag(xh_wT + (size_t)(g * HID + n0) * VOCAB + k, VOCAB, lane);
        acc[g] = wmma_bf16(a, b, acc[g]);
      }
    }
  }
  // h_{t-1} contribution, K = HID = 512 (skipped at t==0 since h0 == 0)
  if (hprev) {
    const u16* Hrow = hprev + (size_t)m0 * HID;
    for (int k = 0; k < HID; k += 32) {
      v16bf a = load_frag(Hrow + k, HID, lane);
      #pragma unroll
      for (int g = 0; g < 4; ++g) {
        v16bf b = load_frag(hh_wT + (size_t)(g * HID + n0) * HID + k, HID, lane);
        acc[g] = wmma_bf16(a, b, acc[g]);
      }
    }
  }

  const int col = lane & 15, hi = lane >> 4;
  float gb[4];
  #pragma unroll
  for (int g = 0; g < 4; ++g)
    gb[g] = xh_b[g * HID + n0 + col] + hh_b[g * HID + n0 + col];

  #pragma unroll
  for (int r = 0; r < 8; ++r) {
    const int row = m0 + 8 * hi + r;
    const size_t idx = (size_t)row * HID + n0 + col;
    float gi = acc[0][r] + gb[0];
    float gf = acc[1][r] + gb[1];
    float gg = acc[2][r] + gb[2];
    float go = acc[3][r] + gb[3];
    if (eimg) {                       // t == 0: gates += concat(e,e,e,e)
      float ev = eimg[idx];
      gi += ev; gf += ev; gg += ev; go += ev;
    }
    float i = sigmoidf(gi);
    float f = sigmoidf(gf);
    float g = tanhf(gg);
    float o = sigmoidf(go);
    float cp = hprev ? cstate[idx] : 0.0f;
    float cn = fmaf(f, cp, i * g);
    cstate[idx] = cn;
    hout[idx] = f2bf(o * tanhf(cn));
  }
}

// ---------------------------------------------------------------------------
// Output GEMM + log_softmax: logits = hs[TB][H] @ out_wT[V][H]^T + b, then
// log_softmax over V=128. One wave per 16-row block covers all 8 N tiles.
// ---------------------------------------------------------------------------
__global__ void __launch_bounds__(32)
k_out_logsoftmax(const u16* __restrict__ hs, const u16* __restrict__ Wt,
                 const float* __restrict__ bias, float* __restrict__ out) {
  __shared__ float sm[16][VOCAB + 1];
  const int lane = threadIdx.x;
  const int m0 = blockIdx.x * 16;       // row block in [T*B]
  const u16* Arow = hs + (size_t)m0 * HID;

  v8f acc[8];
  #pragma unroll
  for (int n = 0; n < 8; ++n) acc[n] = vzero8();
  for (int k = 0; k < HID; k += 32) {
    v16bf a = load_frag(Arow + k, HID, lane);
    #pragma unroll
    for (int n = 0; n < 8; ++n) {
      v16bf b = load_frag(Wt + (size_t)(n * 16) * HID + k, HID, lane);
      acc[n] = wmma_bf16(a, b, acc[n]);
    }
  }
  const int col = lane & 15, hi = lane >> 4;
  #pragma unroll
  for (int n = 0; n < 8; ++n) {
    float bv = bias[n * 16 + col];
    #pragma unroll
    for (int r = 0; r < 8; ++r)
      sm[8 * hi + r][n * 16 + col] = acc[n][r] + bv;
  }
  __syncthreads();
  if (lane < 16) {
    const int row = lane;
    float mx = -3.4e38f;
    for (int j = 0; j < VOCAB; ++j) mx = fmaxf(mx, sm[row][j]);
    float s = 0.0f;
    for (int j = 0; j < VOCAB; ++j) s += expf(sm[row][j] - mx);
    float lse = mx + logf(s);
    float* op = out + (size_t)(m0 + row) * VOCAB;
    for (int j = 0; j < VOCAB; ++j) op[j] = sm[row][j] - lse;
  }
}

// ---------------------------------------------------------------------------
// Host: workspace carving + launch sequence (all on `stream`)
// ---------------------------------------------------------------------------
static inline void* carve(char*& p, size_t bytes) {
  void* r = (void*)p;
  p += (bytes + 255) & ~(size_t)255;
  return r;
}

extern "C" void kernel_launch(void* const* d_in, const int* in_sizes, int n_in,
                              void* d_out, int out_size, void* d_ws, size_t ws_size,
                              hipStream_t stream) {
  (void)in_sizes; (void)n_in; (void)out_size; (void)ws_size;
  const float* inp  = (const float*)d_in[0];   // [T,B,V]
  const float* img  = (const float*)d_in[1];   // [B,1,64,64]
  const float* c1w  = (const float*)d_in[2];
  const float* c1b  = (const float*)d_in[3];
  const float* c2w  = (const float*)d_in[4];
  const float* c2b  = (const float*)d_in[5];
  const float* fcw  = (const float*)d_in[6];   // [FEAT, HID]
  const float* fcb  = (const float*)d_in[7];
  const float* xhw  = (const float*)d_in[8];   // [V, 4H]
  const float* xhb  = (const float*)d_in[9];
  const float* hhw  = (const float*)d_in[10];  // [H, 4H]
  const float* hhb  = (const float*)d_in[11];
  const float* outw = (const float*)d_in[12];  // [H, V]
  const float* outb = (const float*)d_in[13];
  float* out = (float*)d_out;                  // [T,B,V] log-probs

  char* p = (char*)d_ws;
  u16* inp_bf  = (u16*)carve(p, (size_t)T_LEN * BATCH * VOCAB * 2);
  u16* xh_wT   = (u16*)carve(p, (size_t)G4 * VOCAB * 2);
  u16* hh_wT   = (u16*)carve(p, (size_t)G4 * HID * 2);
  u16* out_wT  = (u16*)carve(p, (size_t)VOCAB * HID * 2);
  u16* fc_wT   = (u16*)carve(p, (size_t)HID * FEATP * 2);
  float* pool1 = (float*)carve(p, (size_t)BATCH * 8 * 32 * 32 * 4);
  u16* feat    = (u16*)carve(p, (size_t)BATCH * FEATP * 2);
  float* eimg  = (float*)carve(p, (size_t)BATCH * HID * 4);
  float* cst   = (float*)carve(p, (size_t)BATCH * HID * 4);
  u16* hs      = (u16*)carve(p, (size_t)T_LEN * BATCH * HID * 2);

  const int TB = 256;
  // --- precision prep: bf16 conversions / weight transposes ---
  {
    int n = T_LEN * BATCH * VOCAB;
    k_convert_rows<<<(n + TB - 1) / TB, TB, 0, stream>>>(inp, inp_bf, n);
  }
  { int n = G4 * VOCAB;  k_transpose_convert<<<(n + TB - 1) / TB, TB, 0, stream>>>(xhw,  xh_wT,  VOCAB, G4,    VOCAB); }
  { int n = G4 * HID;    k_transpose_convert<<<(n + TB - 1) / TB, TB, 0, stream>>>(hhw,  hh_wT,  HID,   G4,    HID); }
  { int n = VOCAB * HID; k_transpose_convert<<<(n + TB - 1) / TB, TB, 0, stream>>>(outw, out_wT, HID,   VOCAB, HID); }
  { int n = HID * FEATP; k_transpose_convert<<<(n + TB - 1) / TB, TB, 0, stream>>>(fcw,  fc_wT,  FEAT,  HID,   FEATP); }

  // --- CNN image embedding ---
  { int n = BATCH * 8 * 32 * 32;  k_conv1_pool<<<(n + TB - 1) / TB, TB, 0, stream>>>(img, c1w, c1b, pool1); }
  { int n = BATCH * 16 * 15 * 15; k_conv2_pool_flatten<<<(n + TB - 1) / TB, TB, 0, stream>>>(pool1, c2w, c2b, feat); }
  { int n = BATCH * (FEATP - FEAT); k_zero_featpad<<<(n + TB - 1) / TB, TB, 0, stream>>>(feat); }
  k_imgfc_gemm<<<dim3(HID / 16, BATCH / 16), 32, 0, stream>>>(feat, fc_wT, fcb, eimg);

  // --- LSTM: 64 fused GEMM+cell steps (hh GEMM K=512 dominates; xg fused in) ---
  for (int t = 0; t < T_LEN; ++t) {
    const u16* xt = inp_bf + (size_t)t * BATCH * VOCAB;
    const u16* hp = t ? hs + (size_t)(t - 1) * BATCH * HID : nullptr;
    u16* ho = hs + (size_t)t * BATCH * HID;
    k_lstm_step<<<dim3(HID / 16, BATCH / 16), 32, 0, stream>>>(
        xt, xh_wT, xhb, hp, hh_wT, hhb, (t == 0) ? eimg : nullptr, cst, ho);
  }

  // --- output projection + log_softmax ---
  k_out_logsoftmax<<<dim3((T_LEN * BATCH) / 16), 32, 0, stream>>>(hs, out_wT, outb, out);
}
